// Gate_45861660787104
// MI455X (gfx1250) — matrix operand order repaired and checked by reading, not compiled, not run on hardware
//
#include <hip/hip_runtime.h>
#include <hip/hip_bf16.h>

#define BB 8
#define NN 4096
#define CC 1024
#define HH 16
#define DD 64
#define ATT_SCALE 0.125f   // D^-0.5 = 64^-0.5
#define OSTR 36            // LDS row stride (floats): 16B-aligned, bank-safe

typedef __attribute__((ext_vector_type(16))) __bf16 v16bf;
typedef __attribute__((ext_vector_type(8)))  __bf16 v8bf;
typedef __attribute__((ext_vector_type(8)))  float  v8f;
typedef __attribute__((ext_vector_type(4)))  float  v4f;

// ---------------------------------------------------------------------------
// Kernel 1: k[b,c] = emb[b,:]·Wk[c,:] ; v[b,c] = emb[b,:]·Wv[c,:]
// ---------------------------------------------------------------------------
__global__ __launch_bounds__(256) void gate_kv_kernel(
    const float* __restrict__ emb, const float* __restrict__ Wk,
    const float* __restrict__ Wv, float* __restrict__ k, float* __restrict__ v) {
  const int b = blockIdx.x;
  const int c = blockIdx.y * 256 + threadIdx.x;
  const float* e   = emb + (size_t)b * CC;
  const float* wkr = Wk + (size_t)c * CC;
  const float* wvr = Wv + (size_t)c * CC;
  float aK = 0.f, aV = 0.f;
#pragma unroll 8
  for (int i = 0; i < CC; ++i) {
    float ev = e[i];
    aK += ev * wkr[i];
    aV += ev * wvr[i];
  }
  k[b * CC + c] = aK;
  v[b * CC + c] = aV;
}

// ---------------------------------------------------------------------------
// Kernel 2: fold k into Wq and v into Wp (KV seq len == 1 algebraic collapse)
//   wk16[b][h][c] = sum_d Wq[h*64+d, c] * k[b, h*64+d]     (bf16, c-contig)
//   wp16[b][c][h] = sum_d Wp[c, h*64+d] * v[b, h*64+d]     (bf16, h-contig)
// ---------------------------------------------------------------------------
__global__ __launch_bounds__(256) void gate_eff_kernel(
    const float* __restrict__ Wq, const float* __restrict__ Wp,
    const float* __restrict__ k, const float* __restrict__ v,
    __bf16* __restrict__ wk16, __bf16* __restrict__ wp16) {
  const int b = blockIdx.x, h = blockIdx.y;
  const int c = blockIdx.z * 256 + threadIdx.x;

  float aQ = 0.f;
#pragma unroll 8
  for (int d = 0; d < DD; ++d)
    aQ += Wq[(size_t)(h * DD + d) * CC + c] * k[b * CC + h * DD + d];
  wk16[((size_t)b * HH + h) * CC + c] = (__bf16)aQ;

  float aP = 0.f;
  const float* wpr = Wp + (size_t)c * CC + h * DD;
  const float* vr  = v + b * CC + h * DD;
#pragma unroll 8
  for (int d = 0; d < DD; ++d) aP += wpr[d] * vr[d];
  wp16[((size_t)b * CC + c) * HH + h] = (__bf16)aP;
}

// ---------------------------------------------------------------------------
// Kernel 3 (fused attn + output), block 256 = 8 waves x 16 rows, grid (B,N/128)
// Phase 1: attn[16n x 16h] = sigmoid(SCALE * fea[16n,1024] @ wk_eff[1024,16])
//          via v_wmma_f32_16x16x32_bf16, B staged in 32KB LDS.
// Phase 2: per 32-column strip: 2 WMMAs (K=16 padded to 32, A loop-invariant),
//          D-tiles swizzled through wave-private LDS so the residual load,
//          bias add, and output store run as 128B-contiguous b128 ops.
//          fea residual re-read hits L2 (streamed in phase 1; in-flight
//          working set << 192MB L2).
// ---------------------------------------------------------------------------
__global__ __launch_bounds__(256) void gate_fused_kernel(
    const float* __restrict__ fea, const __bf16* __restrict__ wk16,
    const __bf16* __restrict__ wp16, const float* __restrict__ bp,
    float* __restrict__ out) {
  __shared__ __align__(16) __bf16 lds_wk[HH * CC];        // 32 KB
  __shared__ __align__(16) __bf16 lds_attn[8 * 16 * HH];  //  4 KB
  __shared__ __align__(16) float  lds_out[8 * 16 * OSTR]; // 18 KB

  const int b = blockIdx.x;
  const int tid = threadIdx.x;

  // cooperative 32KB LDS fill of wk_eff[b]: 2048 x 16B chunks / 256 threads
  {
    const uint4* src = (const uint4*)(wk16 + (size_t)b * HH * CC);
    uint4* dst = (uint4*)lds_wk;
#pragma unroll
    for (int i = 0; i < 8; ++i) dst[tid + i * 256] = src[tid + i * 256];
  }
  __syncthreads();

  const int wave = tid >> 5;
  const int lane = tid & 31;
  const int l = lane & 15;        // row (A) / column (B,D) index
  const int hi = lane >> 4;       // lane half selects K slice
  const int n0 = blockIdx.y * 128 + wave * 16;

  const float* fr = fea + ((size_t)b * NN + (n0 + l)) * CC;

  // ---- Phase 1: attention logits [16 x 16] over K = 1024 -----------------
  v8f acc = {};
  for (int kb = 0; kb < CC; kb += 32) {
    // A tile (fea rows, f32 -> bf16). 16-bit A 16x32 layout:
    // elements 0..7 -> K = kb + hi*8 + e ; elements 8..15 -> K = kb+16+hi*8+e
    const int base1 = kb + hi * 8;
    v4f f0 = *(const v4f*)(fr + base1);
    v4f f1 = *(const v4f*)(fr + base1 + 4);
    v4f f2 = *(const v4f*)(fr + base1 + 16);
    v4f f3 = *(const v4f*)(fr + base1 + 20);
    v16bf a;
    a[0]  = (__bf16)f0[0]; a[1]  = (__bf16)f0[1]; a[2]  = (__bf16)f0[2]; a[3]  = (__bf16)f0[3];
    a[4]  = (__bf16)f1[0]; a[5]  = (__bf16)f1[1]; a[6]  = (__bf16)f1[2]; a[7]  = (__bf16)f1[3];
    a[8]  = (__bf16)f2[0]; a[9]  = (__bf16)f2[1]; a[10] = (__bf16)f2[2]; a[11] = (__bf16)f2[3];
    a[12] = (__bf16)f3[0]; a[13] = (__bf16)f3[1]; a[14] = (__bf16)f3[2]; a[15] = (__bf16)f3[3];

    // B tile (wk_eff column l). 16-bit B 32x16 layout:
    // elements 0..7 -> K = kb + hi*8 + e ; elements 8..15 -> K = kb+16+hi*8+e
    const __bf16* wr = lds_wk + l * CC + kb + hi * 8;
    v8bf w0 = *(const v8bf*)(wr);
    v8bf w1 = *(const v8bf*)(wr + 16);
    v16bf bv = __builtin_shufflevector(w0, w1, 0, 1, 2, 3, 4, 5, 6, 7,
                                       8, 9, 10, 11, 12, 13, 14, 15);

    acc = __builtin_amdgcn_wmma_f32_16x16x32_bf16(
        false, a, false, bv, (short)0, acc, false, false);
  }

  // sigmoid via v_exp_f32 + v_rcp_f32; store into wave-private LDS in
  // [row][h] (A-operand) order.
  __bf16* at = lds_attn + wave * 16 * HH;
#pragma unroll
  for (int r = 0; r < 8; ++r) {
    const float x = acc[r] * ATT_SCALE;
    const float s = __builtin_amdgcn_rcpf(1.0f + __expf(-x));
    at[(r + 8 * hi) * HH + l] = (__bf16)s;  // D: VGPR r -> row r+8*hi, col l
  }
  __syncthreads();

  // ---- Phase 2 ------------------------------------------------------------
  // A (loop-invariant): attn rows, K(h)=0..15 live, 16..31 zero.
  v8bf alo = *(const v8bf*)(at + l * HH);
  v8bf zero8 = *(const v8bf*)(at + l * HH + 8);
  v16bf av = __builtin_shufflevector(alo, zero8, 0, 1, 2, 3, 4, 5, 6, 7,
                                     8, 9, 10, 11, 12, 13, 14, 15);
  if (hi) av = (v16bf){};

  float* ow = lds_out + wave * 16 * OSTR;
  const int colb = (lane & 7) * 4;  // 4 consecutive c per lane in gather
  const int rowb = lane >> 3;       // 0..3

  for (int ct = 0; ct < CC / 32; ++ct) {
    const int c0 = ct * 32;

    // Two 16x16 WMMA tiles covering columns [c0, c0+32).
    // B: column c holds K = hi*16 + e -> h; only lanes 0..15 (K=0..15) live.
    v16bf bv0 = {}, bv1 = {};
    if (hi == 0) {
      const v8bf* wp0 = (const v8bf*)(wp16 + ((size_t)b * CC + c0 + l) * HH);
      const v8bf* wp1 = (const v8bf*)(wp16 + ((size_t)b * CC + c0 + 16 + l) * HH);
      bv0 = __builtin_shufflevector(wp0[0], wp0[1], 0, 1, 2, 3, 4, 5, 6, 7,
                                    8, 9, 10, 11, 12, 13, 14, 15);
      bv1 = __builtin_shufflevector(wp1[0], wp1[1], 0, 1, 2, 3, 4, 5, 6, 7,
                                    8, 9, 10, 11, 12, 13, 14, 15);
    }
    v8f z = {};
    v8f oacc0 = __builtin_amdgcn_wmma_f32_16x16x32_bf16(
        false, av, false, bv0, (short)0, z, false, false);
    v8f oacc1 = __builtin_amdgcn_wmma_f32_16x16x32_bf16(
        false, av, false, bv1, (short)0, z, false, false);

    // Scatter D layout (row = r+8*hi, col = l [+16]) into padded LDS strip.
    // Bank-safe: lane-half row offset 8*OSTR = 288 -> +32 banks.
#pragma unroll
    for (int r = 0; r < 8; ++r) {
      ow[(r + 8 * hi) * OSTR + l]      = oacc0[r];
      ow[(r + 8 * hi) * OSTR + l + 16] = oacc1[r];
    }
    // Gather 4 consecutive c per lane; wave-local LDS is in-order (DScnt),
    // no barrier needed. Residual + bias fused into 128B b128 transactions.
    v4f bpv = *(const v4f*)(bp + c0 + colb);
#pragma unroll
    for (int j = 0; j < 4; ++j) {
      const int row = rowb + 4 * j;
      v4f res = *(const v4f*)(ow + row * OSTR + colb);
      const size_t idx = ((size_t)b * NN + (n0 + row)) * CC + c0 + colb;
      v4f fv = *(const v4f*)(fea + idx);
      v4f o = res + fv + bpv;
      *(v4f*)(out + idx) = o;
    }
  }
}

// ---------------------------------------------------------------------------
extern "C" void kernel_launch(void* const* d_in, const int* in_sizes, int n_in,
                              void* d_out, int out_size, void* d_ws, size_t ws_size,
                              hipStream_t stream) {
  const float* fea = (const float*)d_in[0];
  const float* emb = (const float*)d_in[1];
  const float* Wq  = (const float*)d_in[2];
  const float* Wk  = (const float*)d_in[3];
  const float* Wv  = (const float*)d_in[4];
  const float* Wp  = (const float*)d_in[5];
  const float* bp  = (const float*)d_in[6];
  float* out = (float*)d_out;

  char* ws = (char*)d_ws;
  float*  k    = (float*)(ws);                   //  32 KB
  float*  v    = (float*)(ws + 32768);           //  32 KB
  __bf16* wk16 = (__bf16*)(ws + 65536);          // 256 KB [b][h][c]
  __bf16* wp16 = (__bf16*)(ws + 65536 + 262144); // 256 KB [b][c][h]

  gate_kv_kernel<<<dim3(BB, CC / 256), 256, 0, stream>>>(emb, Wk, Wv, k, v);
  gate_eff_kernel<<<dim3(BB, HH, CC / 256), 256, 0, stream>>>(Wq, Wp, k, v,
                                                              wk16, wp16);
  gate_fused_kernel<<<dim3(BB, NN / 128), 256, 0, stream>>>(fea, wk16, wp16,
                                                            bp, out);
}
